// CLIPVisionModelForPromptTuning_41893111005535
// MI455X (gfx1250) — matrix operand order repaired and checked by reading, not compiled, not run on hardware
//
#include <hip/hip_runtime.h>
#include <hip/hip_bf16.h>

typedef __attribute__((ext_vector_type(16))) _Float16 v16h;
typedef __attribute__((ext_vector_type(8)))  float    v8f;

#define TB 256
#define BM 128
#define BN 128
#define BK 32

static constexpr int D_   = 768;
static constexpr int H_   = 12;
static constexpr int DH_  = 64;
static constexpr int L_   = 12;
static constexpr int FF_  = 3072;
static constexpr int P_   = 16;
static constexpr int IMG_ = 224;
static constexpr int NPR_ = 8;
static constexpr int NP_  = 196;          // 14*14 patches
static constexpr int S_   = 205;          // 1 + 196 + 8 tokens
static constexpr int B_   = 64;
static constexpr int MS_  = B_ * S_;      // 13120 rows
static constexpr int MP_  = B_ * NP_;     // 12544 patch rows

// ---------------------------------------------------------------------------
// Generic batched GEMM:  C[z] = act( (A[z] * B[z]^T + bias) * alpha ) + res
//   A: [M,K] row-major (lda), batch offset = (z/div)*sA1 + (z%div)*sA2
//   B: b_kn==0 -> [N,K] row-major (ldb)   (weights / K-matrix)
//      b_kn==1 -> [K,N] row-major (ldb)   (V-matrix; transposed during staging)
//   C: [M,N] (ldc), res shares C's offsets/ldc.
// ---------------------------------------------------------------------------
struct GemmP {
    const float* A; const float* Bm; float* C;
    const float* bias; const float* res;
    long long lda, ldb, ldc;
    long long sA1, sA2, sB1, sB2, sC1, sC2;
    int div;
    int M, N, K;
    float alpha;
    int act;    // 0 = none, 1 = QuickGELU
    int b_kn;
};

__device__ __forceinline__ v16h frag_a(const _Float16* row, int kSel) {
    // ISA 16-bit A layout: lanes 0-15 hold K in {0..7, 16..23}, lanes 16-31 {8..15, 24..31}
    union { v16h v; uint4 u[2]; } r;
    r.u[0] = *reinterpret_cast<const uint4*>(row + kSel * 8);
    r.u[1] = *reinterpret_cast<const uint4*>(row + 16 + kSel * 8);
    return r.v;
}

__device__ __forceinline__ v16h frag_b(const _Float16* row, int kSel) {
    // ISA B layout: lanes 0-15 hold K in {0..7, 16..23}, lanes 16-31 {8..15, 24..31}
    union { v16h v; uint4 u[2]; } r;
    r.u[0] = *reinterpret_cast<const uint4*>(row + kSel * 8);
    r.u[1] = *reinterpret_cast<const uint4*>(row + 16 + kSel * 8);
    return r.v;
}

__global__ __launch_bounds__(TB) void gemm_wmma(GemmP p) {
    __shared__ _Float16 As[BM][BK];
    __shared__ _Float16 Bs[BN][BK];

    const int z  = blockIdx.z;
    const int zb = z / p.div, zh = z % p.div;
    const float* A  = p.A  + (long long)zb * p.sA1 + (long long)zh * p.sA2;
    const float* Bm = p.Bm + (long long)zb * p.sB1 + (long long)zh * p.sB2;
    float*       C  = p.C  + (long long)zb * p.sC1 + (long long)zh * p.sC2;
    const float* R  = p.res ? (p.res + (long long)zb * p.sC1 + (long long)zh * p.sC2) : nullptr;

    const int m0   = blockIdx.x * BM;
    const int n0   = blockIdx.y * BN;
    const int tid  = threadIdx.x;
    const int wave = tid >> 5;
    const int lane = tid & 31;
    const int wm   = (wave >> 2) * 64;    // wave M offset: 0 / 64
    const int wn   = (wave & 3) * 32;     // wave N offset: 0/32/64/96
    const int lrow = lane & 15;
    const int kSel = lane >> 4;

    const bool fastA = ((p.lda & 3) == 0);
    const bool fastB = (p.b_kn == 0) && ((p.ldb & 3) == 0);

    // Prefetch registers: 4 float4 slots per thread per tile (128x32 / 256 thr).
    float4 aReg[4], bReg[4];

    // ---- staging loaders: issue all global loads (unrolled, independent) ----
    auto loadA = [&](int k0) {
#pragma unroll
        for (int i = 0; i < 4; ++i) {
            int s = tid + i * TB;           // float4 slot id 0..1023
            int r = s >> 3, c = (s & 7) * 4;
            int gm = m0 + r, gk = k0 + c;
            float4 v = make_float4(0.f, 0.f, 0.f, 0.f);
            if (gm < p.M) {
                const float* src = A + (long long)gm * p.lda + gk;
                if (fastA && gk + 3 < p.K) {
                    v = *reinterpret_cast<const float4*>(src);
                } else {
                    float* pv = reinterpret_cast<float*>(&v);
#pragma unroll
                    for (int e = 0; e < 4; ++e)
                        if (gk + e < p.K) pv[e] = src[e];
                }
            }
            aReg[i] = v;
        }
    };
    auto loadB = [&](int k0) {
        if (p.b_kn == 0) {
#pragma unroll
            for (int i = 0; i < 4; ++i) {
                int s = tid + i * TB;
                int r = s >> 3, c = (s & 7) * 4;
                int gn = n0 + r, gk = k0 + c;
                float4 v = make_float4(0.f, 0.f, 0.f, 0.f);
                if (gn < p.N) {
                    const float* src = Bm + (long long)gn * p.ldb + gk;
                    if (fastB && gk + 3 < p.K) {
                        v = *reinterpret_cast<const float4*>(src);
                    } else {
                        float* pv = reinterpret_cast<float*>(&v);
#pragma unroll
                        for (int e = 0; e < 4; ++e)
                            if (gk + e < p.K) pv[e] = src[e];
                    }
                }
                bReg[i] = v;
            }
        } else {
            // B is [K,N]: gather 16 scalars, coalesced over n (s = k*128 + n)
            float* pv = reinterpret_cast<float*>(bReg);
#pragma unroll
            for (int i = 0; i < 16; ++i) {
                int s = tid + i * TB;       // element id 0..4095
                int k = s >> 7, n = s & 127;
                int gn = n0 + n, gk = k0 + k;
                float v = 0.f;
                if (gn < p.N && gk < p.K) v = Bm[(long long)gk * p.ldb + gn];
                pv[i] = v;
            }
        }
    };
    // ---- staging stores: convert fp32 -> f16, packed 8-byte LDS writes ----
    auto storeA = [&]() {
#pragma unroll
        for (int i = 0; i < 4; ++i) {
            int s = tid + i * TB;
            int r = s >> 3, c = (s & 7) * 4;
            union { uint2 u; _Float16 h[4]; } pk;
            const float* f = reinterpret_cast<const float*>(&aReg[i]);
#pragma unroll
            for (int e = 0; e < 4; ++e) pk.h[e] = (_Float16)f[e];
            *reinterpret_cast<uint2*>(&As[r][c]) = pk.u;
        }
    };
    auto storeB = [&]() {
        if (p.b_kn == 0) {
#pragma unroll
            for (int i = 0; i < 4; ++i) {
                int s = tid + i * TB;
                int r = s >> 3, c = (s & 7) * 4;
                union { uint2 u; _Float16 h[4]; } pk;
                const float* f = reinterpret_cast<const float*>(&bReg[i]);
#pragma unroll
                for (int e = 0; e < 4; ++e) pk.h[e] = (_Float16)f[e];
                *reinterpret_cast<uint2*>(&Bs[r][c]) = pk.u;
            }
        } else {
            const float* pv = reinterpret_cast<const float*>(bReg);
#pragma unroll
            for (int i = 0; i < 16; ++i) {
                int s = tid + i * TB;
                int k = s >> 7, n = s & 127;
                Bs[n][k] = (_Float16)pv[i];
            }
        }
    };

    v8f acc[4][2];
#pragma unroll
    for (int i = 0; i < 4; ++i)
#pragma unroll
        for (int j = 0; j < 2; ++j) acc[i][j] = v8f{};

    // ---- software-pipelined main loop: prefetch tile k+1 during tile k math ----
    loadA(0);
    loadB(0);
    for (int k0 = 0; k0 < p.K; k0 += BK) {
        storeA();
        storeB();
        __syncthreads();

        if (k0 + BK < p.K) {            // issue next tile's global loads early
            loadA(k0 + BK);
            loadB(k0 + BK);
        }

        v16h af[4], bf[2];
#pragma unroll
        for (int i = 0; i < 4; ++i) af[i] = frag_a(&As[wm + i * 16 + lrow][0], kSel);
#pragma unroll
        for (int j = 0; j < 2; ++j) bf[j] = frag_b(&Bs[wn + j * 16 + lrow][0], kSel);

#pragma unroll
        for (int i = 0; i < 4; ++i)
#pragma unroll
            for (int j = 0; j < 2; ++j)
                acc[i][j] = __builtin_amdgcn_wmma_f32_16x16x32_f16(
                    false, af[i], false, bf[j], (short)0, acc[i][j], false, false);
        __syncthreads();
    }

    // ---- epilogue: C/D layout -> lane 0-15: N=lane, M=r; lane 16-31: M=8+r ----
    const int nIdx = lane & 15;
    const int mOff = (lane >> 4) * 8;
#pragma unroll
    for (int i = 0; i < 4; ++i) {
#pragma unroll
        for (int j = 0; j < 2; ++j) {
            int gn = n0 + wn + j * 16 + nIdx;
#pragma unroll
            for (int r = 0; r < 8; ++r) {
                int gm = m0 + wm + i * 16 + mOff + r;
                if (gm < p.M && gn < p.N) {
                    float v = acc[i][j][r];
                    if (p.bias) v += p.bias[gn];
                    v *= p.alpha;
                    if (p.act == 1) v = v * (1.f / (1.f + __expf(-1.702f * v)));
                    long long off = (long long)gm * p.ldc + gn;
                    if (R) v += R[off];
                    C[off] = v;
                }
            }
        }
    }
}

// ---------------------------------------------------------------------------
// Wave-per-row LayerNorm over rows of width 768 (= 24 x 32 lanes).
// ---------------------------------------------------------------------------
__global__ __launch_bounds__(256) void layernorm_rows(
    const float* __restrict__ x, long long xstr,
    const float* __restrict__ g, const float* __restrict__ b,
    float* __restrict__ out, long long ostr, int nrows) {
    int row  = blockIdx.x * 8 + (threadIdx.x >> 5);
    int lane = threadIdx.x & 31;
    if (row >= nrows) return;
    const float* xr = x + (long long)row * xstr;
    float v[24];
    float s = 0.f;
#pragma unroll
    for (int i = 0; i < 24; ++i) { v[i] = xr[lane + 32 * i]; s += v[i]; }
#pragma unroll
    for (int o = 16; o > 0; o >>= 1) s += __shfl_xor(s, o, 32);
    float mu = s * (1.f / 768.f);
    float q = 0.f;
#pragma unroll
    for (int i = 0; i < 24; ++i) { float d = v[i] - mu; q += d * d; }
#pragma unroll
    for (int o = 16; o > 0; o >>= 1) q += __shfl_xor(q, o, 32);
    float rstd = rsqrtf(q * (1.f / 768.f) + 1e-5f);
    float* orow = out + (long long)row * ostr;
#pragma unroll
    for (int i = 0; i < 24; ++i) {
        int c = lane + 32 * i;
        orow[c] = (v[i] - mu) * rstd * g[c] + b[c];
    }
}

// ---------------------------------------------------------------------------
// Wave-per-row softmax over rows of length 205 (<= 7 x 32).
// ---------------------------------------------------------------------------
__global__ __launch_bounds__(256) void softmax_rows(float* __restrict__ x, int nrows) {
    int row  = blockIdx.x * 8 + (threadIdx.x >> 5);
    int lane = threadIdx.x & 31;
    if (row >= nrows) return;
    float* xr = x + (long long)row * S_;
    float v[7];
    float mx = -3.0e38f;
#pragma unroll
    for (int i = 0; i < 7; ++i) {
        int c = lane + 32 * i;
        v[i] = (c < S_) ? xr[c] : -3.0e38f;
        mx = fmaxf(mx, v[i]);
    }
#pragma unroll
    for (int o = 16; o > 0; o >>= 1) mx = fmaxf(mx, __shfl_xor(mx, o, 32));
    float s = 0.f;
#pragma unroll
    for (int i = 0; i < 7; ++i) {
        int c = lane + 32 * i;
        float e = (c < S_) ? __expf(v[i] - mx) : 0.f;
        v[i] = e; s += e;
    }
#pragma unroll
    for (int o = 16; o > 0; o >>= 1) s += __shfl_xor(s, o, 32);
    float inv = 1.f / s;
#pragma unroll
    for (int i = 0; i < 7; ++i) {
        int c = lane + 32 * i;
        if (c < S_) xr[c] = v[i] * inv;
    }
}

// ---------------------------------------------------------------------------
// im2col for the 16x16/stride-16 patch conv: out[b*196+p][c*256+ph*16+pw]
// ---------------------------------------------------------------------------
__global__ __launch_bounds__(256) void im2col_patches(const float* __restrict__ img,
                                                      float* __restrict__ out) {
    long long idx = (long long)blockIdx.x * 256 + threadIdx.x;
    const long long total = (long long)MP_ * D_;
    if (idx >= total) return;
    int col = (int)(idx % D_);
    long long rowi = idx / D_;
    int pch = (int)(rowi % NP_);
    int b   = (int)(rowi / NP_);
    int c  = col >> 8;
    int ph = (col >> 4) & 15;
    int pw = col & 15;
    int py = pch / 14, px = pch % 14;
    out[idx] = img[(((long long)b * 3 + c) * IMG_ + (py * P_ + ph)) * IMG_ + (px * P_ + pw)];
}

// ---------------------------------------------------------------------------
// Build x = [cls | patches | prompts(layer 0)] + pos (pos only on first 197)
// ---------------------------------------------------------------------------
__global__ __launch_bounds__(256) void build_embed(const float* __restrict__ feat,
                                                   const float* __restrict__ cls,
                                                   const float* __restrict__ pos,
                                                   const float* __restrict__ gp,
                                                   float* __restrict__ x) {
    long long idx = (long long)blockIdx.x * 256 + threadIdx.x;
    const long long total = (long long)MS_ * D_;
    if (idx >= total) return;
    int d = (int)(idx % D_);
    long long rowi = idx / D_;
    int s = (int)(rowi % S_);
    int b = (int)(rowi / S_);
    float v;
    if (s == 0) {
        v = cls[d] + pos[d];
    } else if (s <= NP_) {
        v = feat[((long long)b * NP_ + (s - 1)) * D_ + d] + pos[(long long)s * D_ + d];
    } else {
        v = gp[(((long long)b * L_ + 0) * NPR_ + (s - 1 - NP_)) * D_ + d];
    }
    x[idx] = v;
}

// ---------------------------------------------------------------------------
// Replace last NPR tokens with layer-i prompts
// ---------------------------------------------------------------------------
__global__ __launch_bounds__(256) void replace_prompts(const float* __restrict__ gp, int layer,
                                                       float* __restrict__ x) {
    long long idx = (long long)blockIdx.x * 256 + threadIdx.x;
    const long long total = (long long)B_ * NPR_ * D_;
    if (idx >= total) return;
    int d = (int)(idx % D_);
    long long r = idx / D_;
    int j = (int)(r % NPR_);
    int b = (int)(r / NPR_);
    x[((long long)b * S_ + (S_ - NPR_) + j) * D_ + d] =
        gp[(((long long)b * L_ + layer) * NPR_ + j) * D_ + d];
}

// ---------------------------------------------------------------------------
// Host-side launch helpers
// ---------------------------------------------------------------------------
static void run_gemm(hipStream_t st,
                     const float* A, long long lda, long long sA1, long long sA2,
                     const float* Bm, long long ldb, long long sB1, long long sB2, int b_kn,
                     float* C, long long ldc, long long sC1, long long sC2,
                     const float* bias, const float* res,
                     int M, int N, int K, int batches, int div, float alpha, int act) {
    GemmP p;
    p.A = A; p.Bm = Bm; p.C = C; p.bias = bias; p.res = res;
    p.lda = lda; p.ldb = ldb; p.ldc = ldc;
    p.sA1 = sA1; p.sA2 = sA2; p.sB1 = sB1; p.sB2 = sB2; p.sC1 = sC1; p.sC2 = sC2;
    p.div = div; p.M = M; p.N = N; p.K = K; p.alpha = alpha; p.act = act; p.b_kn = b_kn;
    dim3 grid((M + BM - 1) / BM, (N + BN - 1) / BN, batches);
    hipLaunchKernelGGL(gemm_wmma, grid, dim3(TB), 0, st, p);
}

extern "C" void kernel_launch(void* const* d_in, const int* in_sizes, int n_in,
                              void* d_out, int out_size, void* d_ws, size_t ws_size,
                              hipStream_t stream) {
    const float* image   = (const float*)d_in[0];
    const float* gp      = (const float*)d_in[1];
    const float* patch_w = (const float*)d_in[2];
    const float* cls     = (const float*)d_in[3];
    const float* pos     = (const float*)d_in[4];
    const float* pre_g   = (const float*)d_in[5];
    const float* pre_b   = (const float*)d_in[6];
    const float* ln1_g   = (const float*)d_in[7];
    const float* ln1_b   = (const float*)d_in[8];
    const float* qw      = (const float*)d_in[9];
    const float* qb      = (const float*)d_in[10];
    const float* kw      = (const float*)d_in[11];
    const float* kb      = (const float*)d_in[12];
    const float* vw      = (const float*)d_in[13];
    const float* vb      = (const float*)d_in[14];
    const float* ow      = (const float*)d_in[15];
    const float* ob      = (const float*)d_in[16];
    const float* ln2_g   = (const float*)d_in[17];
    const float* ln2_b   = (const float*)d_in[18];
    const float* fc1_w   = (const float*)d_in[19];
    const float* fc1_b   = (const float*)d_in[20];
    const float* fc2_w   = (const float*)d_in[21];
    const float* fc2_b   = (const float*)d_in[22];
    const float* post_g  = (const float*)d_in[23];
    const float* post_b  = (const float*)d_in[24];
    float* out = (float*)d_out;

    // -------- workspace carving (floats) --------
    float* ws = (float*)d_ws;
    size_t o = 0;
    float* xbuf   = ws + o; o += (size_t)MS_ * D_;
    float* hbuf   = ws + o; o += (size_t)MS_ * D_;
    float* qbuf   = ws + o; o += (size_t)MS_ * D_;
    float* kbuf   = ws + o; o += (size_t)MS_ * D_;
    float* vbuf   = ws + o; o += (size_t)MS_ * D_;
    float* abuf   = ws + o; o += (size_t)MS_ * D_;
    float* scores = ws + o; o += (size_t)B_ * H_ * S_ * S_;
    float* hidden = ws + o; o += (size_t)MS_ * FF_;
    float* imcol  = hidden;     // alias: only used before MLP buffer is needed
    float* feat   = abuf;       // alias: only used before attention buffer is needed

    const long long sdD  = (long long)S_ * D_;      // per-image stride in token buffers
    const long long ssBH = (long long)S_ * S_;      // per-(b,h) stride in scores

    // -------- patch embedding: im2col + GEMM vs patch_w [768 x 768] --------
    {
        long long tot = (long long)MP_ * D_;
        im2col_patches<<<dim3((unsigned)((tot + 255) / 256)), dim3(256), 0, stream>>>(image, imcol);
        run_gemm(stream, imcol, D_, 0, 0, patch_w, D_, 0, 0, 0,
                 feat, D_, 0, 0, nullptr, nullptr, MP_, D_, D_, 1, 1, 1.f, 0);
        long long te = (long long)MS_ * D_;
        build_embed<<<dim3((unsigned)((te + 255) / 256)), dim3(256), 0, stream>>>(
            feat, cls, pos, gp, xbuf);
        layernorm_rows<<<dim3((MS_ + 7) / 8), dim3(256), 0, stream>>>(
            xbuf, D_, pre_g, pre_b, xbuf, D_, MS_);
    }

    // -------- transformer layers --------
    for (int i = 0; i < L_; ++i) {
        if (i > 0) {  // DEEP_G = 12 -> replace prompts at every layer 1..11
            long long tp = (long long)B_ * NPR_ * D_;
            replace_prompts<<<dim3((unsigned)((tp + 255) / 256)), dim3(256), 0, stream>>>(
                gp, i, xbuf);
        }
        const size_t wOff = (size_t)i * D_ * D_;

        // LN1
        layernorm_rows<<<dim3((MS_ + 7) / 8), dim3(256), 0, stream>>>(
            xbuf, D_, ln1_g + (size_t)i * D_, ln1_b + (size_t)i * D_, hbuf, D_, MS_);

        // Q (scaled), K, V projections
        run_gemm(stream, hbuf, D_, 0, 0, qw + wOff, D_, 0, 0, 0,
                 qbuf, D_, 0, 0, qb + (size_t)i * D_, nullptr,
                 MS_, D_, D_, 1, 1, 0.125f, 0);
        run_gemm(stream, hbuf, D_, 0, 0, kw + wOff, D_, 0, 0, 0,
                 kbuf, D_, 0, 0, kb + (size_t)i * D_, nullptr,
                 MS_, D_, D_, 1, 1, 1.f, 0);
        run_gemm(stream, hbuf, D_, 0, 0, vw + wOff, D_, 0, 0, 0,
                 vbuf, D_, 0, 0, vb + (size_t)i * D_, nullptr,
                 MS_, D_, D_, 1, 1, 1.f, 0);

        // scores[b,h] = Q[b,h] (S x 64) * K[b,h]^T  — batched over 768 (b,h)
        run_gemm(stream,
                 qbuf, D_, sdD, DH_,
                 kbuf, D_, sdD, DH_, 0,
                 scores, S_, (long long)H_ * ssBH, ssBH,
                 nullptr, nullptr, S_, S_, DH_, B_ * H_, H_, 1.f, 0);

        softmax_rows<<<dim3((B_ * H_ * S_ + 7) / 8), dim3(256), 0, stream>>>(
            scores, B_ * H_ * S_);

        // attn_out[b,h] = scores[b,h] (S x S) * V[b,h] (S x 64)  (V staged transposed)
        run_gemm(stream,
                 scores, S_, (long long)H_ * ssBH, ssBH,
                 vbuf, D_, sdD, DH_, 1,
                 abuf, D_, sdD, DH_,
                 nullptr, nullptr, S_, DH_, S_, B_ * H_, H_, 1.f, 0);

        // output projection + residual
        run_gemm(stream, abuf, D_, 0, 0, ow + wOff, D_, 0, 0, 0,
                 xbuf, D_, 0, 0, ob + (size_t)i * D_, xbuf,
                 MS_, D_, D_, 1, 1, 1.f, 0);

        // LN2
        layernorm_rows<<<dim3((MS_ + 7) / 8), dim3(256), 0, stream>>>(
            xbuf, D_, ln2_g + (size_t)i * D_, ln2_b + (size_t)i * D_, hbuf, D_, MS_);

        // MLP: fc1 + QuickGELU, fc2 + residual
        run_gemm(stream, hbuf, D_, 0, 0, fc1_w + (size_t)i * FF_ * D_, D_, 0, 0, 0,
                 hidden, FF_, 0, 0, fc1_b + (size_t)i * FF_, nullptr,
                 MS_, FF_, D_, 1, 1, 1.f, 1);
        run_gemm(stream, hidden, FF_, 0, 0, fc2_w + (size_t)i * D_ * FF_, FF_, 0, 0, 0,
                 xbuf, D_, 0, 0, fc2_b + (size_t)i * D_, xbuf,
                 MS_, D_, FF_, 1, 1, 1.f, 0);
    }

    // -------- final LN on CLS token of each image --------
    layernorm_rows<<<dim3((B_ + 7) / 8), dim3(256), 0, stream>>>(
        xbuf, (long long)S_ * D_, post_g, post_b, out, D_, B_);
}
